// FEMRMamba_67903432950544
// MI455X (gfx1250) — matrix operand order repaired and checked by reading, not compiled, not run on hardware
//
#include <hip/hip_runtime.h>
#include <hip/hip_bf16.h>
#include <math.h>

// ---------------- problem constants ----------------
#define VOC   32768
#define DM    768
#define DII   1536
#define NST   16
#define RKC   48
#define NLC   2
#define SSEG  4
#define LSEG  2048
#define TTOT  (SSEG*LSEG)      // 8192 tokens
#define DCONV 4
#define EPSR  1e-5f
#define SSMW  (RKC + 2*NST)    // 80
#define NCH   16               // scan chunks per segment
#define CL    (LSEG/NCH)       // 128 steps per chunk

typedef __attribute__((ext_vector_type(16))) __bf16 v16bf;
typedef __attribute__((ext_vector_type(8)))  float  v8f;

// =====================================================================
// GEMM: C[M x Nn] = A[M x K] * B[Nn x K]^T  (all f32 in memory)
// bf16x3 split (hi+lo) -> ~fp32 accuracy on v_wmma_f32_16x16x32_bf16.
// Each wave computes 1 M-tile x NT N-tiles, reusing the A fragment.
// mode 0: store; mode 1: softplus(x + bias[col]); mode 2: C += result
// =====================================================================
template<bool KALIGNED, int NT>
__global__ __launch_bounds__(256) void gemm_bf16x3(
    const float* __restrict__ A, int lda,
    const float* __restrict__ B, int ldb,
    float* __restrict__ C, int ldc,
    int M, int Nn, int K,
    int mode, const float* __restrict__ bias)
{
    const int lane  = threadIdx.x & 31;
    const int wave  = threadIdx.x >> 5;
    const int tg    = blockIdx.x * 8 + wave;     // N tile-group index
    const int nTiles = Nn >> 4;
    if (tg * NT >= nTiles) return;               // wave-uniform guard
    const int tileM = blockIdx.y;

    const int half  = lane >> 4;                 // which K-half this lane covers
    const int idx16 = lane & 15;

    const float* __restrict__ arow = A + (size_t)(tileM * 16 + idx16) * lda;
    const float* brow[NT];
    #pragma unroll
    for (int j = 0; j < NT; ++j)
        brow[j] = B + (size_t)((tg * NT + j) * 16 + idx16) * ldb;

    v8f acc[NT];
    #pragma unroll
    for (int j = 0; j < NT; ++j) { v8f z = {}; acc[j] = z; }

    for (int k0 = 0; k0 < K; k0 += 32) {
        // ---- A fragment (loaded & split once, reused for NT tiles) ----
        float af[16];
        if (KALIGNED) {
            const float* pa = arow + k0 + half * 8;
            float4 a0 = *(const float4*)(pa);
            float4 a1 = *(const float4*)(pa + 4);
            float4 a2 = *(const float4*)(pa + 16);
            float4 a3 = *(const float4*)(pa + 20);
            af[0]=a0.x; af[1]=a0.y; af[2]=a0.z; af[3]=a0.w;
            af[4]=a1.x; af[5]=a1.y; af[6]=a1.z; af[7]=a1.w;
            af[8]=a2.x; af[9]=a2.y; af[10]=a2.z; af[11]=a2.w;
            af[12]=a3.x; af[13]=a3.y; af[14]=a3.z; af[15]=a3.w;
        } else {
            #pragma unroll
            for (int e = 0; e < 16; ++e) {
                int ka = k0 + (e < 8 ? half * 8 + e : 8 + half * 8 + e);
                af[e] = (ka < K) ? arow[ka] : 0.0f;
            }
        }
        v16bf ah, al;
        #pragma unroll
        for (int e = 0; e < 16; ++e) {
            __bf16 h = (__bf16)af[e];
            ah[e] = h;  al[e] = (__bf16)(af[e] - (float)h);
        }

        // ---- NT B fragments, 3 WMMAs each ----
        #pragma unroll
        for (int j = 0; j < NT; ++j) {
            float bf[16];
            if (KALIGNED) {
                const float* pb = brow[j] + k0 + half * 8;
                float4 b0 = *(const float4*)(pb);
                float4 b1 = *(const float4*)(pb + 4);
                float4 b2 = *(const float4*)(pb + 16);
                float4 b3 = *(const float4*)(pb + 20);
                bf[0]=b0.x; bf[1]=b0.y; bf[2]=b0.z; bf[3]=b0.w;
                bf[4]=b1.x; bf[5]=b1.y; bf[6]=b1.z; bf[7]=b1.w;
                bf[8]=b2.x; bf[9]=b2.y; bf[10]=b2.z; bf[11]=b2.w;
                bf[12]=b3.x; bf[13]=b3.y; bf[14]=b3.z; bf[15]=b3.w;
            } else {
                #pragma unroll
                for (int e = 0; e < 16; ++e) {
                    int kb = k0 + (e < 8 ? half * 8 + e : 8 + half * 8 + e);
                    bf[e] = (kb < K) ? brow[j][kb] : 0.0f;
                }
            }
            v16bf bh, bl;
            #pragma unroll
            for (int e = 0; e < 16; ++e) {
                __bf16 g = (__bf16)bf[e];
                bh[e] = g;  bl[e] = (__bf16)(bf[e] - (float)g);
            }
            acc[j] = __builtin_amdgcn_wmma_f32_16x16x32_bf16(false, ah, false, bh, (short)0, acc[j], false, false);
            acc[j] = __builtin_amdgcn_wmma_f32_16x16x32_bf16(false, ah, false, bl, (short)0, acc[j], false, false);
            acc[j] = __builtin_amdgcn_wmma_f32_16x16x32_bf16(false, al, false, bh, (short)0, acc[j], false, false);
        }
    }

    // D layout: VGPR r holds (M = tileM*16 + half*8 + r, N = tile*16 + idx16)
    const int mbase = tileM * 16 + half * 8;
    #pragma unroll
    for (int j = 0; j < NT; ++j) {
        const int col = (tg * NT + j) * 16 + idx16;
        #pragma unroll
        for (int r = 0; r < 8; ++r) {
            float v = acc[j][r];
            size_t off = (size_t)(mbase + r) * ldc + col;
            if (mode == 1) {
                v += bias[col];
                v = (v > 20.0f) ? v : log1pf(__expf(v));   // softplus
            } else if (mode == 2) {
                v += C[off];                                // residual accumulate
            }
            C[off] = v;
        }
    }
}

// =====================================================================
// Embedding gather + rmsnorm (one block per token row)
// =====================================================================
__global__ __launch_bounds__(256) void embed_rms_kernel(
    const int* __restrict__ tokens, const float* __restrict__ embed,
    const float* __restrict__ w, float* __restrict__ out)
{
    __shared__ float sdata[256];
    const int row = blockIdx.x, tid = threadIdx.x;
    const float* __restrict__ xr = embed + (size_t)tokens[row] * DM;
    float v[3]; float ss = 0.0f;
    #pragma unroll
    for (int i = 0; i < 3; ++i) { v[i] = xr[tid + i * 256]; ss += v[i] * v[i]; }
    sdata[tid] = ss; __syncthreads();
    for (int o = 128; o > 0; o >>= 1) { if (tid < o) sdata[tid] += sdata[tid + o]; __syncthreads(); }
    float r = rsqrtf(sdata[0] / (float)DM + EPSR);
    #pragma unroll
    for (int i = 0; i < 3; ++i) {
        int c = tid + i * 256;
        out[(size_t)row * DM + c] = v[i] * r * w[c];
    }
}

// rmsnorm (one block per row)
__global__ __launch_bounds__(256) void rms_kernel(
    const float* __restrict__ x, const float* __restrict__ w, float* __restrict__ out)
{
    __shared__ float sdata[256];
    const int row = blockIdx.x, tid = threadIdx.x;
    const float* __restrict__ xr = x + (size_t)row * DM;
    float v[3]; float ss = 0.0f;
    #pragma unroll
    for (int i = 0; i < 3; ++i) { v[i] = xr[tid + i * 256]; ss += v[i] * v[i]; }
    sdata[tid] = ss; __syncthreads();
    for (int o = 128; o > 0; o >>= 1) { if (tid < o) sdata[tid] += sdata[tid + o]; __syncthreads(); }
    float r = rsqrtf(sdata[0] / (float)DM + EPSR);
    #pragma unroll
    for (int i = 0; i < 3; ++i) {
        int c = tid + i * 256;
        out[(size_t)row * DM + c] = v[i] * r * w[c];
    }
}

// fused double rmsnorm: rmsnorm(rmsnorm(x, w1), w2)
__global__ __launch_bounds__(256) void final_norm_kernel(
    const float* __restrict__ x, const float* __restrict__ w1,
    const float* __restrict__ w2, float* __restrict__ out)
{
    __shared__ float sdata[256];
    const int row = blockIdx.x, tid = threadIdx.x;
    const float* __restrict__ xr = x + (size_t)row * DM;
    float v[3]; float ss = 0.0f;
    #pragma unroll
    for (int i = 0; i < 3; ++i) { v[i] = xr[tid + i * 256]; ss += v[i] * v[i]; }
    sdata[tid] = ss; __syncthreads();
    for (int o = 128; o > 0; o >>= 1) { if (tid < o) sdata[tid] += sdata[tid + o]; __syncthreads(); }
    float r1 = rsqrtf(sdata[0] / (float)DM + EPSR);
    __syncthreads();
    float u[3]; float ss2 = 0.0f;
    #pragma unroll
    for (int i = 0; i < 3; ++i) {
        int c = tid + i * 256;
        u[i] = v[i] * r1 * w1[c];
        ss2 += u[i] * u[i];
    }
    sdata[tid] = ss2; __syncthreads();
    for (int o = 128; o > 0; o >>= 1) { if (tid < o) sdata[tid] += sdata[tid + o]; __syncthreads(); }
    float r2 = rsqrtf(sdata[0] / (float)DM + EPSR);
    #pragma unroll
    for (int i = 0; i < 3; ++i) {
        int c = tid + i * 256;
        out[(size_t)row * DM + c] = u[i] * r2 * w2[c];
    }
}

// =====================================================================
// Causal depthwise conv (within 2048-token segments) + bias + SiLU.
// h is the low half of hz (row stride 2*DII).
// =====================================================================
__global__ __launch_bounds__(256) void conv_silu_kernel(
    const float* __restrict__ hz, const float* __restrict__ cw,
    const float* __restrict__ cb, float* __restrict__ hc)
{
    size_t gid = (size_t)blockIdx.x * 256 + threadIdx.x;
    if (gid >= (size_t)TTOT * DII) return;
    int d = (int)(gid % DII);
    int row = (int)(gid / DII);
    int t = row % LSEG;
    float acc = cb[d];
    #pragma unroll
    for (int k = 0; k < DCONV; ++k) {
        int tt = t - (DCONV - 1) + k;
        if (tt >= 0) acc += hz[(size_t)(row - (DCONV - 1) + k) * (2 * DII) + d] * cw[d * DCONV + k];
    }
    acc = acc / (1.0f + __expf(-acc));   // SiLU
    hc[gid] = acc;
}

// =====================================================================
// Chunked selective scan (linear recurrence => chunk decomposition).
// Pass 1: per-(seg,chunk,channel) scan from zero state; record end
//         state and sum(dt) (chunk total decay == exp(A * sum dt)).
// =====================================================================
__global__ __launch_bounds__(256) void scan_part1_kernel(
    const float* __restrict__ dt, const float* __restrict__ hc,
    const float* __restrict__ ssm, const float* __restrict__ Alog,
    float* __restrict__ pend, float* __restrict__ sumdt)
{
    int gid = blockIdx.x * 256 + threadIdx.x;
    if (gid >= SSEG * NCH * DII) return;
    int d  = gid % DII;
    int sc = gid / DII;            // s*NCH + c
    int c  = sc % NCH;
    int s  = sc / NCH;

    float negA[NST], st[NST];
    #pragma unroll
    for (int n = 0; n < NST; ++n) { negA[n] = -__expf(Alog[d * NST + n]); st[n] = 0.0f; }

    float sd = 0.0f;
    const int t0 = c * CL;
    for (int t = t0; t < t0 + CL; ++t) {
        size_t row = (size_t)s * LSEG + t;
        float dtv = dt[row * DII + d];
        float xv  = hc[row * DII + d];
        float dx  = dtv * xv;
        const float* __restrict__ bc = ssm + row * SSMW + RKC;
        sd += dtv;
        #pragma unroll
        for (int n = 0; n < NST; ++n)
            st[n] = st[n] * __expf(dtv * negA[n]) + dx * bc[n];
    }
    size_t base = ((size_t)sc * DII + d) * NST;
    #pragma unroll
    for (int n = 0; n < NST; n += 4)
        *(float4*)(pend + base + n) = make_float4(st[n], st[n+1], st[n+2], st[n+3]);
    sumdt[(size_t)sc * DII + d] = sd;
}

// Pass 2: tiny sequential fixup over NCH chunks -> per-chunk init state.
__global__ __launch_bounds__(256) void scan_fixup_kernel(
    const float* __restrict__ pend, const float* __restrict__ sumdt,
    const float* __restrict__ Alog, float* __restrict__ hst)
{
    int gid = blockIdx.x * 256 + threadIdx.x;
    if (gid >= SSEG * DII) return;
    int d = gid % DII;
    int s = gid / DII;

    float negA[NST], H[NST];
    #pragma unroll
    for (int n = 0; n < NST; ++n) { negA[n] = -__expf(Alog[d * NST + n]); H[n] = 0.0f; }

    for (int c = 0; c < NCH; ++c) {
        int sc = s * NCH + c;
        size_t base = ((size_t)sc * DII + d) * NST;
        float sd = sumdt[(size_t)sc * DII + d];
        #pragma unroll
        for (int n = 0; n < NST; ++n) {
            hst[base + n] = H[n];
            H[n] = H[n] * __expf(sd * negA[n]) + pend[base + n];
        }
    }
}

// Pass 3: re-scan each chunk from its true initial state; emit gated y.
__global__ __launch_bounds__(256) void scan_part2_kernel(
    const float* __restrict__ dt, const float* __restrict__ hc,
    const float* __restrict__ ssm, const float* __restrict__ hz,
    const float* __restrict__ Alog, const float* __restrict__ Dskip,
    const float* __restrict__ hst, float* __restrict__ yact)
{
    int gid = blockIdx.x * 256 + threadIdx.x;
    if (gid >= SSEG * NCH * DII) return;
    int d  = gid % DII;
    int sc = gid / DII;
    int c  = sc % NCH;
    int s  = sc / NCH;

    float negA[NST], st[NST];
    size_t base = ((size_t)sc * DII + d) * NST;
    #pragma unroll
    for (int n = 0; n < NST; ++n) {
        negA[n] = -__expf(Alog[d * NST + n]);
        st[n]   = hst[base + n];
    }
    const float dsk = Dskip[d];

    const int t0 = c * CL;
    for (int t = t0; t < t0 + CL; ++t) {
        size_t row = (size_t)s * LSEG + t;
        float dtv = dt[row * DII + d];
        float xv  = hc[row * DII + d];
        float dx  = dtv * xv;
        const float* __restrict__ bc = ssm + row * SSMW + RKC;  // B[0..15], C[16..31]
        float y = 0.0f;
        #pragma unroll
        for (int n = 0; n < NST; ++n) {
            float e = __expf(dtv * negA[n]);
            st[n] = st[n] * e + dx * bc[n];
            y += st[n] * bc[NST + n];
        }
        float zv = hz[row * (2 * DII) + DII + d];
        float g  = zv / (1.0f + __expf(-zv));                   // silu(z)
        yact[row * DII + d] = (y + xv * dsk) * g;
    }
}

// =====================================================================
// Host orchestration
// =====================================================================
extern "C" void kernel_launch(void* const* d_in, const int* in_sizes, int n_in,
                              void* d_out, int out_size, void* d_ws, size_t ws_size,
                              hipStream_t stream) {
    (void)in_sizes; (void)n_in; (void)out_size; (void)ws_size;

    const int*   tokens    = (const int*)  d_in[0];
    // d_in[1] = subject_lengths (all LSEG; unused)
    const float* embed     = (const float*)d_in[2];
    const float* in_norm_w = (const float*)d_in[3];
    const float* out_norm_w= (const float*)d_in[4];
    const float* norm_f_w  = (const float*)d_in[5];
    const float* ln_w      = (const float*)d_in[6];
    const float* in_w      = (const float*)d_in[7];
    const float* conv_w    = (const float*)d_in[8];
    const float* conv_b    = (const float*)d_in[9];
    const float* xp_w      = (const float*)d_in[10];
    const float* dt_w      = (const float*)d_in[11];
    const float* dt_b      = (const float*)d_in[12];
    const float* A_log     = (const float*)d_in[13];
    const float* D_skip    = (const float*)d_in[14];
    const float* out_w     = (const float*)d_in[15];

    // workspace partition (floats)
    float* X     = (float*)d_ws;                      // residual stream 8192x768
    float* XN    = X     + (size_t)TTOT * DM;         // pre-norm         8192x768
    float* HZ    = XN    + (size_t)TTOT * DM;         // in-proj out      8192x3072
    float* HC    = HZ    + (size_t)TTOT * 2 * DII;    // conv out         8192x1536
    float* SSM   = HC    + (size_t)TTOT * DII;        // x-proj out       8192x80
    float* DT    = SSM   + (size_t)TTOT * SSMW;       // dt (softplus)    8192x1536
    float* YACT  = DT    + (size_t)TTOT * DII;        // gated scan out   8192x1536
    float* PEND  = YACT  + (size_t)TTOT * DII;        // chunk end states
    float* SUMDT = PEND  + (size_t)SSEG * NCH * DII * NST;
    float* HST   = SUMDT + (size_t)SSEG * NCH * DII;  // chunk init states

    dim3 blk(256);
    const int chBlocks = (SSEG * NCH * DII) / 256;    // 384

    // 1) embedding gather + input rmsnorm
    embed_rms_kernel<<<TTOT, blk, 0, stream>>>(tokens, embed, in_norm_w, X);

    // 2) two Mamba layers
    for (int l = 0; l < NLC; ++l) {
        const float* in_w_l   = in_w   + (size_t)l * 2 * DII * DM;
        const float* conv_w_l = conv_w + (size_t)l * DII * DCONV;
        const float* conv_b_l = conv_b + (size_t)l * DII;
        const float* xp_w_l   = xp_w   + (size_t)l * SSMW * DII;
        const float* dt_w_l   = dt_w   + (size_t)l * DII * RKC;
        const float* dt_b_l   = dt_b   + (size_t)l * DII;
        const float* A_log_l  = A_log  + (size_t)l * DII * NST;
        const float* D_skip_l = D_skip + (size_t)l * DII;
        const float* out_w_l  = out_w  + (size_t)l * DM * DII;
        const float* ln_w_l   = ln_w   + (size_t)l * DM;

        // pre-norm
        rms_kernel<<<TTOT, blk, 0, stream>>>(X, ln_w_l, XN);

        // in-proj: HZ = XN @ in_w^T   (8192x3072, K=768); 192 tiles / (8*4) = 6
        gemm_bf16x3<true, 4><<<dim3(6, TTOT / 16), blk, 0, stream>>>(
            XN, DM, in_w_l, DM, HZ, 2 * DII, TTOT, 2 * DII, DM, 0, nullptr);

        // depthwise causal conv + bias + SiLU on h-half
        conv_silu_kernel<<<(TTOT * DII) / 256, blk, 0, stream>>>(HZ, conv_w_l, conv_b_l, HC);

        // x-proj: SSM = HC @ xp_w^T   (8192x80, K=1536); 5 tiles -> 1 block, NT=1
        gemm_bf16x3<true, 1><<<dim3(1, TTOT / 16), blk, 0, stream>>>(
            HC, DII, xp_w_l, DII, SSM, SSMW, TTOT, SSMW, DII, 0, nullptr);

        // dt: DT = softplus(SSM[:, :48] @ dt_w^T + dt_b); 96 tiles / 32 = 3
        gemm_bf16x3<false, 4><<<dim3(3, TTOT / 16), blk, 0, stream>>>(
            SSM, SSMW, dt_w_l, RKC, DT, DII, TTOT, DII, RKC, 1, dt_b_l);

        // chunked selective scan + D-skip + z-gating
        scan_part1_kernel<<<chBlocks, blk, 0, stream>>>(DT, HC, SSM, A_log_l, PEND, SUMDT);
        scan_fixup_kernel<<<(SSEG * DII) / 256, blk, 0, stream>>>(PEND, SUMDT, A_log_l, HST);
        scan_part2_kernel<<<chBlocks, blk, 0, stream>>>(DT, HC, SSM, HZ, A_log_l, D_skip_l, HST, YACT);

        // out-proj + residual: X += YACT @ out_w^T; 48 tiles / 16 = 3, NT=2
        gemm_bf16x3<true, 2><<<dim3(3, TTOT / 16), blk, 0, stream>>>(
            YACT, DII, out_w_l, DII, X, DM, TTOT, DM, DII, 2, nullptr);
    }

    // 3) fused rmsnorm(norm_f_w) then rmsnorm(out_norm_w)
    final_norm_kernel<<<TTOT, blk, 0, stream>>>(X, norm_f_w, out_norm_w, (float*)d_out);
}
